// MultiHeadSelfAttentionModule_14585708937733
// MI455X (gfx1250) — matrix operand order repaired and checked by reading, not compiled, not run on hardware
//
#include <hip/hip_runtime.h>
#include <hip/hip_bf16.h>
#include <stddef.h>

// ---------- CDNA5 WMMA types ----------
typedef _Float16 half_t;
typedef __attribute__((ext_vector_type(16))) _Float16 v16h;
typedef __attribute__((ext_vector_type(8)))  _Float16 v8h;
typedef __attribute__((ext_vector_type(8)))  float    v8f;
typedef __attribute__((ext_vector_type(4)))  int      v4i;

#define B_    4
#define L_    2048
#define DM    512
#define H_    8
#define DH    64
#define NPOS_ 100
#define RPAD  208   // 201 rel positions padded to 13*16

#if defined(__has_builtin)
#  if __has_builtin(__builtin_amdgcn_global_load_async_to_lds_b128)
#    define HAVE_ASYNC_LDS 1
#  endif
#endif
#ifndef HAVE_ASYNC_LDS
#  define HAVE_ASYNC_LDS 0
#endif

union AFrag { v16h v; v8h h[2]; };

static __device__ __forceinline__ v8h  ld8h (const half_t* p){ return *reinterpret_cast<const v8h*>(p); }
static __device__ __forceinline__ v16h ld16h(const half_t* p){ return *reinterpret_cast<const v16h*>(p); }
static __device__ __forceinline__ v8f  vzero(){ v8f z = {0.f,0.f,0.f,0.f,0.f,0.f,0.f,0.f}; return z; }
static __device__ __forceinline__ v8f  wmma16(v16h a, v16h b, v8f c){
  // D = A(16x32 f16) * B(32x16 f16) + C(16x16 f32)
  return __builtin_amdgcn_wmma_f32_16x16x32_f16(false, a, false, b, (short)0, c, false, false);
}

// 16-byte global -> LDS copy: async (ASYNCcnt) on CDNA5, sync fallback otherwise.
static __device__ __forceinline__ void cp16_g2l(const half_t* g, half_t* l) {
#if HAVE_ASYNC_LDS
  __builtin_amdgcn_global_load_async_to_lds_b128(
      (__attribute__((address_space(1))) v4i*)g,
      (__attribute__((address_space(3))) v4i*)l, 0, 0);
#else
  *reinterpret_cast<uint4*>(l) = *reinterpret_cast<const uint4*>(g);
#endif
}
static __device__ __forceinline__ void async_wait_all() {
#if HAVE_ASYNC_LDS
  asm volatile("s_wait_asynccnt 0x0" ::: "memory");
#endif
}

// ---------- weight transpose + f16 convert : wT[n*512+d] = w[d*512+n] ----------
__global__ void k_wT(const float* __restrict__ w, half_t* __restrict__ wT) {
  int idx = blockIdx.x * 256 + threadIdx.x;      // 512*512
  int d = idx >> 9, n = idx & 511;
  wT[(size_t)n * DM + d] = (half_t)w[idx];
}

// ---------- rel_emb (201,64) -> f16 padded (208,64) ----------
__global__ void k_rel(const float* __restrict__ rel, half_t* __restrict__ relh) {
  int idx = blockIdx.x * 256 + threadIdx.x;      // 208*64 = 13312 = 52*256
  int r = idx / DH;
  relh[idx] = (r < 2 * NPOS_ + 1) ? (half_t)rel[idx] : (half_t)0.f;
}

// ---------- LayerNorm -> f16 xn ----------
__global__ void k_ln(const float* __restrict__ x, const float* __restrict__ g,
                     const float* __restrict__ be, half_t* __restrict__ xn) {
  int row = blockIdx.x;                          // B*L rows
  int t   = threadIdx.x;                         // 128 threads, 4 elems each
  const float4* xr = reinterpret_cast<const float4*>(x + (size_t)row * DM);
  float4 vv = xr[t];
  float s  = vv.x + vv.y + vv.z + vv.w;
  float s2 = vv.x*vv.x + vv.y*vv.y + vv.z*vv.z + vv.w*vv.w;
  __shared__ float sm[128], sm2[128];
  sm[t] = s; sm2[t] = s2; __syncthreads();
  for (int off = 64; off > 0; off >>= 1) {
    if (t < off) { sm[t] += sm[t+off]; sm2[t] += sm2[t+off]; }
    __syncthreads();
  }
  float mu  = sm[0] * (1.f / DM);
  float var = sm2[0] * (1.f / DM) - mu * mu;
  float rs  = rsqrtf(var + 1e-5f);
  int c = t * 4;
  half_t* outp = xn + (size_t)row * DM + c;
  outp[0] = (half_t)((vv.x - mu) * rs * g[c+0] + be[c+0]);
  outp[1] = (half_t)((vv.y - mu) * rs * g[c+1] + be[c+1]);
  outp[2] = (half_t)((vv.z - mu) * rs * g[c+2] + be[c+2]);
  outp[3] = (half_t)((vv.w - mu) * rs * g[c+3] + be[c+3]);
}

// ---------- QKV projection GEMM (M=8192, N=512, K=512), WMMA ----------
// grid (64, 8, 3) x 256 threads. Wave = 16 rows x 64 cols (4 accs).
// Q,K -> (B,H,L,64) f16 ; V -> transposed (B,H,64,L) f16.
__global__ void k_qkv(const half_t* __restrict__ xn,
                      const half_t* __restrict__ wqT, const half_t* __restrict__ wkT,
                      const half_t* __restrict__ wvT,
                      const float* __restrict__ bq, const float* __restrict__ bk,
                      const float* __restrict__ bv,
                      half_t* __restrict__ qh, half_t* __restrict__ kh,
                      half_t* __restrict__ vT) {
  int wvid = threadIdx.x >> 5;
  int lane = threadIdx.x & 31;
  int l15  = lane & 15, hig = lane >> 4;
  int m0   = blockIdx.x * 128 + wvid * 16;
  int n0   = blockIdx.y * 64;
  int mat  = blockIdx.z;
  const half_t* wT   = (mat == 0) ? wqT : (mat == 1) ? wkT : wvT;
  const float*  bias = (mat == 0) ? bq  : (mat == 1) ? bk  : bv;

  v8f z = vzero();
  v8f acc[4] = {z, z, z, z};
  int row = m0 + l15;
  const half_t* arow = xn + (size_t)row * DM;

  for (int kk = 0; kk < DM; kk += 32) {
    AFrag a;
    a.h[0] = ld8h(arow + kk +     hig * 16);   // K 0..7 (lo) / 16..23 (hi)
    a.h[1] = ld8h(arow + kk + 8 + hig * 16);   // K 8..15     / 24..31
#pragma unroll
    for (int t = 0; t < 4; t++) {
      int n = n0 + t * 16 + l15;
      v16h bf = ld16h(wT + (size_t)n * DM + kk + hig * 16);
      acc[t] = wmma16(a.v, bf, acc[t]);
    }
  }
#pragma unroll
  for (int t = 0; t < 4; t++) {
    int n = n0 + t * 16 + l15;
    float bval = bias[n];
    int h = n >> 6, dh = n & 63;
#pragma unroll
    for (int r = 0; r < 8; r++) {
      int m = m0 + r + 8 * hig;
      int b = m >> 11, l = m & (L_ - 1);
      half_t hv = (half_t)(acc[t][r] + bval);
      if (mat == 2)
        vT[((size_t)(b * H_ + h) * DH + dh) * L_ + l] = hv;
      else if (mat == 0)
        qh[((size_t)(b * H_ + h) * L_ + l) * DH + dh] = hv;
      else
        kh[((size_t)(b * H_ + h) * L_ + l) * DH + dh] = hv;
    }
  }
}

// ---------- qE = Q @ rel_emb^T  (M=B*H*L, N=208, K=64), WMMA ----------
__global__ void k_qE(const half_t* __restrict__ qh, const half_t* __restrict__ relh,
                     half_t* __restrict__ qE) {
  int gw   = (blockIdx.x * blockDim.x + threadIdx.x) >> 5;
  int lane = threadIdx.x & 31;
  int l15  = lane & 15, hig = lane >> 4;
  int mt = gw / 13, rt = gw % 13;
  int row = mt * 16 + l15;
  const half_t* arow = qh + (size_t)row * DH;
  AFrag a0, a1;
  a0.h[0] = ld8h(arow +  0 + hig * 16); a0.h[1] = ld8h(arow +  8 + hig * 16);
  a1.h[0] = ld8h(arow + 32 + hig * 16); a1.h[1] = ld8h(arow + 40 + hig * 16);
  int r = rt * 16 + l15;
  const half_t* brow = relh + (size_t)r * DH;
  v16h b0 = ld16h(brow +      hig * 16);
  v16h b1 = ld16h(brow + 32 + hig * 16);
  v8f acc = wmma16(a0.v, b0, vzero());
  acc      = wmma16(a1.v, b1, acc);
#pragma unroll
  for (int rr = 0; rr < 8; rr++) {
    int m = mt * 16 + rr + 8 * hig;
    qE[(size_t)m * RPAD + r] = (half_t)acc[rr];
  }
}

// ---------- flash attention ----------
// WG = 4 waves = 64 query rows of one (b,h). 64-key K/V blocks are staged in
// LDS cooperatively with double-buffered async global->LDS copies (ASYNCcnt),
// shared by all 4 waves. Per wave: 16 q rows, online softmax, WMMA S and PV.
__global__ void __launch_bounds__(128) k_attn(
    const half_t* __restrict__ qh, const half_t* __restrict__ kh,
    const half_t* __restrict__ vT, const half_t* __restrict__ qE,
    const unsigned char* __restrict__ mask,
    half_t* __restrict__ ofh) {
  __shared__ half_t kbuf[2][64 * 64];   // [key][d]
  __shared__ half_t vbuf[2][64 * 64];   // [d][key]
  __shared__ half_t plds[4][16 * 64];
  int tid  = threadIdx.x;
  int wvid = tid >> 5, lane = tid & 31;
  int l15 = lane & 15, hig = lane >> 4;
  int bh   = blockIdx.x >> 5;           // b*H + h
  int qblk = blockIdx.x & 31;           // 64-row query block
  int b = bh >> 3, h = bh & 7;
  int q0 = qblk * 64 + wvid * 16;
  const half_t* qbase = qh + (size_t)bh * L_ * DH;
  const half_t* kbase = kh + (size_t)bh * L_ * DH;
  const half_t* vbase = vT + (size_t)bh * DH * L_;
  const half_t* qErow = qE + (size_t)bh * L_ * RPAD;
  const unsigned char* mrow = mask + (size_t)b * L_;
  half_t* pl = plds[wvid];

  // load Q A-frags (held in registers for the whole pass)
  int qrow = q0 + l15;
  const half_t* qp = qbase + (size_t)qrow * DH;
  AFrag a0, a1;
  a0.h[0] = ld8h(qp +  0 + hig * 16); a0.h[1] = ld8h(qp +  8 + hig * 16);
  a1.h[0] = ld8h(qp + 32 + hig * 16); a1.h[1] = ld8h(qp + 40 + hig * 16);

  v8f z = vzero();
  v8f accO[4] = {z, z, z, z};
  float mrun[8], lrun[8];
  int   myrow[8];
#pragma unroll
  for (int r = 0; r < 8; r++) { mrun[r] = -1e30f; lrun[r] = 0.f; myrow[r] = q0 + r + 8 * hig; }
  const float scale = 0.125f;   // 64^-0.5

  // cooperative stage of one 64-key block: 128 threads x (4 K + 4 V) b128 chunks
  auto stage = [&](int kb, int buf) {
#pragma unroll
    for (int i = 0; i < 4; i++) {
      int c = tid * 4 + i;                 // 0..511
      int row = c >> 3, part = (c & 7) * 8;
      cp16_g2l(kbase + (size_t)(kb + row) * DH + part, &kbuf[buf][row * 64 + part]);
      cp16_g2l(vbase + (size_t)row * L_ + kb + part,   &vbuf[buf][row * 64 + part]);
    }
  };

  stage(0, 0);
  for (int kb = 0; kb < L_; kb += 64) {
    int cur = (kb >> 6) & 1;
    async_wait_all();          // my async copies into buf[cur] landed
    __syncthreads();           // everyone's did; prior block's readers are done
    if (kb + 64 < L_) stage(kb + 64, cur ^ 1);

    // --- S = Q K^T over 64 keys (LDS K block) ---
    v8f sac[4];
#pragma unroll
    for (int t = 0; t < 4; t++) {
      const half_t* kp = &kbuf[cur][(t * 16 + l15) * 64];
      v16h blo = ld16h(kp +      hig * 16);
      v16h bhi = ld16h(kp + 32 + hig * 16);
      v8f s = wmma16(a0.v, blo, z);
      sac[t] = wmma16(a1.v, bhi, s);
    }
    // --- rel logits gather + mask + scale (C layout: col=lane, row=vgpr) ---
#pragma unroll
    for (int t = 0; t < 4; t++) {
      int n = kb + t * 16 + l15;
      bool msk = mrow[n] != 0;
#pragma unroll
      for (int r = 0; r < 8; r++) {
        int m = myrow[r];
        int d = n - m;
        d = d < -NPOS_ ? -NPOS_ : (d > NPOS_ ? NPOS_ : d);
        float rel = (float)qErow[(size_t)m * RPAD + (d + NPOS_)];
        float val = (sac[t][r] + rel) * scale;
        sac[t][r] = msk ? -1e9f : val;
      }
    }
    // --- online softmax (row stats via width-16 shuffles) ---
    float alpha[8];
#pragma unroll
    for (int r = 0; r < 8; r++) {
      float mb = fmaxf(fmaxf(sac[0][r], sac[1][r]), fmaxf(sac[2][r], sac[3][r]));
      for (int off = 8; off > 0; off >>= 1) mb = fmaxf(mb, __shfl_xor(mb, off, 16));
      float mnew = fmaxf(mrun[r], mb);
      alpha[r] = __expf(mrun[r] - mnew);
      mrun[r] = mnew;
    }
#pragma unroll
    for (int r = 0; r < 8; r++) {
      float rs = 0.f;
#pragma unroll
      for (int t = 0; t < 4; t++) {
        float p = __expf(sac[t][r] - mrun[r]);
        sac[t][r] = p;
        rs += p;
      }
      for (int off = 8; off > 0; off >>= 1) rs += __shfl_xor(rs, off, 16);
      lrun[r] = lrun[r] * alpha[r] + rs;
    }
    // --- P: C layout -> LDS -> A layout (per-wave buffer) ---
#pragma unroll
    for (int t = 0; t < 4; t++)
#pragma unroll
      for (int r = 0; r < 8; r++)
        pl[(r + 8 * hig) * 64 + t * 16 + l15] = (half_t)sac[t][r];
#pragma unroll
    for (int t = 0; t < 4; t++)
#pragma unroll
      for (int r = 0; r < 8; r++)
        accO[t][r] *= alpha[r];
    asm volatile("s_wait_dscnt 0x0" ::: "memory");
    const half_t* pr = pl + (size_t)l15 * 64;
    AFrag pa0, pa1;
    pa0.h[0] = ld8h(pr +  0 + hig * 16); pa0.h[1] = ld8h(pr +  8 + hig * 16);
    pa1.h[0] = ld8h(pr + 32 + hig * 16); pa1.h[1] = ld8h(pr + 40 + hig * 16);
    // --- O += P @ V  (LDS V block: row d, keys contiguous) ---
#pragma unroll
    for (int t = 0; t < 4; t++) {
      const half_t* vp = &vbuf[cur][(t * 16 + l15) * 64];
      v16h vb0 = ld16h(vp +      hig * 16);
      v16h vb1 = ld16h(vp + 32 + hig * 16);
      accO[t] = wmma16(pa0.v, vb0, accO[t]);
      accO[t] = wmma16(pa1.v, vb1, accO[t]);
    }
  }
  // --- epilogue: normalize, store f16 o (B,L,D) head-major cols ---
#pragma unroll
  for (int r = 0; r < 8; r++) {
    float inv = 1.f / lrun[r];
    int m = myrow[r];
    size_t obase = ((size_t)b * L_ + m) * DM + h * DH;
#pragma unroll
    for (int t = 0; t < 4; t++)
      ofh[obase + t * 16 + l15] = (half_t)(accO[t][r] * inv);
  }
}

// ---------- output projection GEMM (M=8192,N=512,K=512) -> fp32 out ----------
__global__ void k_oproj(const half_t* __restrict__ ofh, const half_t* __restrict__ woT,
                        const float* __restrict__ bo, float* __restrict__ out) {
  int wvid = threadIdx.x >> 5;
  int lane = threadIdx.x & 31;
  int l15 = lane & 15, hig = lane >> 4;
  int m0 = blockIdx.x * 128 + wvid * 16;
  int n0 = blockIdx.y * 64;
  v8f z = vzero();
  v8f acc[4] = {z, z, z, z};
  const half_t* arow = ofh + (size_t)(m0 + l15) * DM;
  for (int kk = 0; kk < DM; kk += 32) {
    AFrag a;
    a.h[0] = ld8h(arow + kk +     hig * 16);
    a.h[1] = ld8h(arow + kk + 8 + hig * 16);
#pragma unroll
    for (int t = 0; t < 4; t++) {
      int n = n0 + t * 16 + l15;
      v16h bf = ld16h(woT + (size_t)n * DM + kk + hig * 16);
      acc[t] = wmma16(a.v, bf, acc[t]);
    }
  }
#pragma unroll
  for (int t = 0; t < 4; t++) {
    int n = n0 + t * 16 + l15;
    float bval = bo[n];
#pragma unroll
    for (int r = 0; r < 8; r++) {
      int m = m0 + r + 8 * hig;
      out[(size_t)m * DM + n] = acc[t][r] + bval;
    }
  }
}

// ---------- launch ----------
extern "C" void kernel_launch(void* const* d_in, const int* in_sizes, int n_in,
                              void* d_out, int out_size, void* d_ws, size_t ws_size,
                              hipStream_t stream) {
  const float* x    = (const float*)d_in[0];
  const unsigned char* mask = (const unsigned char*)d_in[1];  // bool mask
  const float* ln_g = (const float*)d_in[2];
  const float* ln_b = (const float*)d_in[3];
  const float* w_q  = (const float*)d_in[4];
  const float* b_q  = (const float*)d_in[5];
  const float* w_k  = (const float*)d_in[6];
  const float* b_k  = (const float*)d_in[7];
  const float* w_v  = (const float*)d_in[8];
  const float* b_v  = (const float*)d_in[9];
  const float* w_o  = (const float*)d_in[10];
  const float* b_o  = (const float*)d_in[11];
  const float* rel  = (const float*)d_in[12];
  float* out = (float*)d_out;

  char* ws = (char*)d_ws;
  const size_t SZ_XN  = (size_t)B_ * L_ * DM * sizeof(half_t);          // 8 MB
  const size_t SZ_QKV = (size_t)B_ * H_ * L_ * DH * sizeof(half_t);     // 8 MB
  const size_t SZ_QE  = (size_t)B_ * H_ * L_ * RPAD * sizeof(half_t);   // ~26 MB
  const size_t SZ_W   = (size_t)DM * DM * sizeof(half_t);               // 0.5 MB
  size_t off = 0;
  half_t* xn  = (half_t*)(ws + off); off += SZ_XN;
  half_t* qh  = (half_t*)(ws + off); off += SZ_QKV;
  half_t* kh  = (half_t*)(ws + off); off += SZ_QKV;
  half_t* vT  = (half_t*)(ws + off); off += SZ_QKV;
  half_t* qE  = (half_t*)(ws + off); off += SZ_QE;
  half_t* ofh = (half_t*)(ws + off); off += SZ_XN;
  half_t* wqT = (half_t*)(ws + off); off += SZ_W;
  half_t* wkT = (half_t*)(ws + off); off += SZ_W;
  half_t* wvT = (half_t*)(ws + off); off += SZ_W;
  half_t* woT = (half_t*)(ws + off); off += SZ_W;
  half_t* relh= (half_t*)(ws + off); off += (size_t)RPAD * DH * sizeof(half_t);

  k_wT <<<1024, 256, 0, stream>>>(w_q, wqT);
  k_wT <<<1024, 256, 0, stream>>>(w_k, wkT);
  k_wT <<<1024, 256, 0, stream>>>(w_v, wvT);
  k_wT <<<1024, 256, 0, stream>>>(w_o, woT);
  k_rel<<<52,   256, 0, stream>>>(rel, relh);
  k_ln <<<B_ * L_, 128, 0, stream>>>(x, ln_g, ln_b, xn);
  k_qkv<<<dim3(64, 8, 3), 256, 0, stream>>>(xn, wqT, wkT, wvT, b_q, b_k, b_v, qh, kh, vT);
  k_qE <<<6656, 256, 0, stream>>>(qh, relh, qE);
  k_attn<<<1024, 128, 0, stream>>>(qh, kh, vT, qE, mask, ofh);
  k_oproj<<<dim3(64, 8), 256, 0, stream>>>(ofh, woT, b_o, out);
}
